// ReMoKE_88751204204866
// MI455X (gfx1250) — compile-verified, matrix operand order checked by reading, not run on hardware
//
#include <hip/hip_runtime.h>
#include <hip/hip_bf16.h>
#include <math.h>

typedef __attribute__((ext_vector_type(16))) __bf16 v16bf;
typedef __attribute__((ext_vector_type(8)))  float  v8f;
typedef __attribute__((ext_vector_type(4)))  __bf16 bf16x4;

#define NT   16384
#define DIN  1024
#define DOUT 1024
#define NE   8
#define A_STR 40   // padded LDS row stride (elements) to spread banks

// ---- CDNA5 async global->LDS copy (16B per lane), tracked by ASYNCcnt ------
__device__ __forceinline__ void async_copy16(unsigned lds_byte_addr,
                                             const __bf16* gaddr) {
    // VDST = per-lane LDS byte address, VADDR = per-lane 64-bit global address
    asm volatile("global_load_async_to_lds_b128 %0, %1, off"
                 :: "v"(lds_byte_addr), "v"(gaddr)
                 : "memory");
}
#define WAIT_ASYNC(n) asm volatile("s_wait_asynccnt " #n ::: "memory")

// low 32 bits of a generic pointer to __shared__ = LDS byte offset
__device__ __forceinline__ unsigned lds_addr_of(const void* p) {
    return (unsigned)(unsigned long long)p;
}

// ---------------- f32 -> bf16 pack (embs then W, contiguous dst) -------------
__global__ void cvt_bf16_kernel(const float* __restrict__ embs,
                                const float* __restrict__ W,
                                __bf16* __restrict__ dst,
                                long long embsN, long long totalN) {
    long long stride = (long long)gridDim.x * blockDim.x;
    for (long long i4 = (long long)blockIdx.x * blockDim.x + threadIdx.x;
         i4 * 4 < totalN; i4 += stride) {
        long long idx = i4 * 4;
        const float* s = (idx < embsN) ? (embs + idx) : (W + (idx - embsN));
        float4 f = *(const float4*)s;
        bf16x4 v;
        v.x = (__bf16)f.x; v.y = (__bf16)f.y; v.z = (__bf16)f.z; v.w = (__bf16)f.w;
        *(bf16x4*)(dst + idx) = v;
    }
}

// ---------------- gating: softmax(embs @ exps_T), one wave per token ---------
__global__ void gate_kernel(const float* __restrict__ embs,
                            const float* __restrict__ expsT,
                            float* __restrict__ weights) {
    int wave = threadIdx.x >> 5;
    int lane = threadIdx.x & 31;
    int n = blockIdx.x * 8 + wave;
    float acc[NE];
#pragma unroll
    for (int e = 0; e < NE; ++e) acc[e] = 0.f;
    for (int d = lane; d < DIN; d += 32) {
        float x = embs[(size_t)n * DIN + d];
        const float4* p = (const float4*)(expsT + (size_t)d * NE);
        float4 a = p[0], b = p[1];
        acc[0] += x * a.x; acc[1] += x * a.y; acc[2] += x * a.z; acc[3] += x * a.w;
        acc[4] += x * b.x; acc[5] += x * b.y; acc[6] += x * b.z; acc[7] += x * b.w;
    }
#pragma unroll
    for (int e = 0; e < NE; ++e)
#pragma unroll
        for (int off = 16; off >= 1; off >>= 1)
            acc[e] += __shfl_xor(acc[e], off, 32);
    if (lane == 0) {
        float m = acc[0];
#pragma unroll
        for (int e = 1; e < NE; ++e) m = fmaxf(m, acc[e]);
        float s = 0.f, ex[NE];
#pragma unroll
        for (int e = 0; e < NE; ++e) { ex[e] = __expf(acc[e] - m); s += ex[e]; }
        float inv = 1.f / s;
#pragma unroll
        for (int e = 0; e < NE; ++e) weights[(size_t)n * NE + e] = ex[e] * inv;
    }
}

// ---------------- correction[e][o] = dot(bias_e, W_e[o,:]) (f32) -------------
__global__ void corr_kernel(const float* __restrict__ bias,
                            const float* __restrict__ W,
                            float* __restrict__ corr) {
    int wave = threadIdx.x >> 5;
    int lane = threadIdx.x & 31;
    int id = blockIdx.x * 8 + wave;          // 0 .. NE*DOUT-1
    int e = id >> 10, o = id & 1023;
    const float* b = bias + (size_t)e * DIN;
    const float* w = W + ((size_t)e * DOUT + o) * DIN;
    float acc = 0.f;
    for (int d = lane; d < DIN; d += 32) acc = fmaf(b[d], w[d], acc);
#pragma unroll
    for (int off = 16; off >= 1; off >>= 1) acc += __shfl_xor(acc, off, 32);
    if (lane == 0) corr[id] = acc;
}

// ---------------- main WMMA kernel: all 8 experts register-resident ----------
// Wave tile: 2 M-tiles x 1 N-tile (32 tokens x 16 outputs) so each B fragment
// feeds two consecutive WMMAs -> half the ds_load pressure, built-in slack.
__launch_bounds__(256)
__global__ void moe_wmma_kernel(const __bf16* __restrict__ Abf,
                                const __bf16* __restrict__ Wbf,
                                const float* __restrict__ weights,
                                const float* __restrict__ corr,
                                float* __restrict__ multi,
                                float* __restrict__ expo) {
    __shared__ __bf16 Abuf[2][64 * A_STR];
    __shared__ __bf16 Wbuf[2][NE][64 * A_STR];
    __shared__ float  Gw[64 * NE];

    int t    = threadIdx.x;
    int wave = t >> 5, lane = t & 31;
    int wn4 = wave & 3;         // n group: 16 cols
    int wm2 = wave >> 2;        // m group: 32 rows
    int tokBase = blockIdx.y * 64;
    int oBase   = blockIdx.x * 64;

    // stage gate weights for this token tile (64 tokens x 8 experts)
    Gw[t]       = weights[(size_t)tokBase * NE + t];
    Gw[t + 256] = weights[(size_t)tokBase * NE + 256 + t];

    v8f acc[NE][2];             // [expert][m-tile]
    {
        v8f z = {};
#pragma unroll
        for (int e = 0; e < NE; ++e) { acc[e][0] = z; acc[e][1] = z; }
    }

    // staging slot for this thread: 16B of one row
    int row = t >> 2;            // 0..63
    int kp  = (t & 3) * 8;       // 0,8,16,24 (elements)
    const __bf16* aSrc = Abf + (size_t)(tokBase + row) * DIN + kp;
    const __bf16* wSrc = Wbf + (size_t)(oBase + row) * DIN + kp;   // + e*DOUT*DIN
    unsigned ldsA[2], ldsW[2];
#pragma unroll
    for (int b = 0; b < 2; ++b) {
        ldsA[b] = lds_addr_of(&Abuf[b][row * A_STR + kp]);
        ldsW[b] = lds_addr_of(&Wbuf[b][0][row * A_STR + kp]);
    }
    const unsigned W_EXP_STRIDE = 64 * A_STR * sizeof(__bf16);   // bytes per expert tile

    // fragment coordinates
    int nl = lane & 15;
    int h  = lane >> 4;
    int am0 = wm2 * 32 + nl;          // A m-tile 0 row
    int am1 = am0 + 16;               // A m-tile 1 row
    int ocol = wn4 * 16 + nl;         // B column row-in-W

    // ---- prologue: issue chunk 0 into buffer 0 ----
    async_copy16(ldsA[0], aSrc);
#pragma unroll
    for (int e = 0; e < NE; ++e)
        async_copy16(ldsW[0] + e * W_EXP_STRIDE, wSrc + (size_t)e * DOUT * DIN);

    union frag { v16bf v; uint4 q[2]; };

    for (int ki = 0; ki < DIN / 32; ++ki) {
        int kc = ki * 32;
        __syncthreads();   // everyone done reading buffer (ki+1)&1 from iter ki-1
        if (ki + 1 < DIN / 32) {
            int b = (ki + 1) & 1;
            const __bf16* an = aSrc + kc + 32;
            const __bf16* wn2 = wSrc + kc + 32;
            async_copy16(ldsA[b], an);
#pragma unroll
            for (int e = 0; e < NE; ++e)
                async_copy16(ldsW[b] + e * W_EXP_STRIDE, wn2 + (size_t)e * DOUT * DIN);
            if (kc + 64 < DIN) {   // L2 prefetch two chunks ahead
                __builtin_prefetch(aSrc + kc + 64, 0, 3);
                __builtin_prefetch(wSrc + kc + 64, 0, 3);
            }
            WAIT_ASYNC(0x9);       // chunk ki's 9 loads (in-order) have landed
        } else {
            WAIT_ASYNC(0x0);
        }
        __syncthreads();           // chunk ki visible to all waves

        int buf = ki & 1;
        const __bf16* As = Abuf[buf];
        const __bf16* Wse[NE];
#pragma unroll
        for (int e = 0; e < NE; ++e) Wse[e] = Wbuf[buf][e];

        // A fragments: lane holds row m, two 16B runs at k = h*8 and 16 + h*8
        frag af0, af1;
        af0.q[0] = *(const uint4*)(As + am0 * A_STR + h * 8);
        af0.q[1] = *(const uint4*)(As + am0 * A_STR + 16 + h * 8);
        af1.q[0] = *(const uint4*)(As + am1 * A_STR + h * 8);
        af1.q[1] = *(const uint4*)(As + am1 * A_STR + 16 + h * 8);

        // ---- pipeline over 8 experts: 1 B-frag load-pair feeds 2 WMMAs ----
        frag bfr[3];
#pragma unroll
        for (int p = 0; p < 2; ++p) {   // preload experts 0,1
            bfr[p].q[0] = *(const uint4*)(Wse[p] + ocol * A_STR + h * 8);
            bfr[p].q[1] = *(const uint4*)(Wse[p] + ocol * A_STR + 16 + h * 8);
        }
#pragma unroll
        for (int e = 0; e < NE; ++e) {
            if (e + 2 < NE) {           // issue loads 2 experts ahead
                bfr[(e + 2) % 3].q[0] =
                    *(const uint4*)(Wse[e + 2] + ocol * A_STR + h * 8);
                bfr[(e + 2) % 3].q[1] =
                    *(const uint4*)(Wse[e + 2] + ocol * A_STR + 16 + h * 8);
            }
            acc[e][0] = __builtin_amdgcn_wmma_f32_16x16x32_bf16(
                false, af0.v, false, bfr[e % 3].v, (short)0, acc[e][0], false, false);
            acc[e][1] = __builtin_amdgcn_wmma_f32_16x16x32_bf16(
                false, af1.v, false, bfr[e % 3].v, (short)0, acc[e][1], false, false);
        }
    }

    // ---- epilogue: subtract correction, store expert_outputs, weighted sum ----
    // C/D layout: lane L, VGPR r  ->  n = L%16, tokenRow = r + 8*(L/16)
    int nGlob = oBase + wn4 * 16 + nl;
#pragma unroll
    for (int mt = 0; mt < 2; ++mt) {
        v8f macc = {};
#pragma unroll
        for (int e = 0; e < NE; ++e) {
            float c = corr[e * DOUT + nGlob];
#pragma unroll
            for (int r = 0; r < 8; ++r) {
                int tokLoc = wm2 * 32 + mt * 16 + h * 8 + r;
                float val = acc[e][mt][r] - c;
                // write-once stream: non-temporal, keep L2 for bf16 A/W tiles
                __builtin_nontemporal_store(
                    val, &expo[((size_t)(tokBase + tokLoc) * NE + e) * DOUT + nGlob]);
                macc[r] += Gw[tokLoc * NE + e] * val;
            }
        }
#pragma unroll
        for (int r = 0; r < 8; ++r) {
            int tokLoc = wm2 * 32 + mt * 16 + h * 8 + r;
            __builtin_nontemporal_store(
                macc[r], &multi[(size_t)(tokBase + tokLoc) * DOUT + nGlob]);
        }
    }
}

// -----------------------------------------------------------------------------
extern "C" void kernel_launch(void* const* d_in, const int* in_sizes, int n_in,
                              void* d_out, int out_size, void* d_ws, size_t ws_size,
                              hipStream_t stream) {
    const float* embs  = (const float*)d_in[0];   // [NT, DIN]
    const float* expsT = (const float*)d_in[1];   // [DIN, NE]
    const float* bias  = (const float*)d_in[2];   // [NE, DIN]
    const float* W     = (const float*)d_in[3];   // [NE, DOUT, DIN]

    float* out   = (float*)d_out;
    float* multi = out;                                  // [NT, DOUT]
    float* expo  = out + (size_t)NT * DOUT;              // [NT, NE, DOUT]

    // workspace layout
    __bf16* Abf = (__bf16*)d_ws;                         // NT*DIN bf16  (32 MB)
    __bf16* Wbf = Abf + (size_t)NT * DIN;                // NE*DOUT*DIN bf16 (16 MB)
    float* weights = (float*)(Wbf + (size_t)NE * DOUT * DIN);  // NT*NE f32
    float* corr    = weights + (size_t)NT * NE;                // NE*DOUT f32

    long long embsN  = (long long)NT * DIN;
    long long totalN = embsN + (long long)NE * DOUT * DIN;
    cvt_bf16_kernel<<<4096, 256, 0, stream>>>(embs, W, Abf, embsN, totalN);
    gate_kernel<<<NT / 8, 256, 0, stream>>>(embs, expsT, weights);
    corr_kernel<<<NE * DOUT / 8, 256, 0, stream>>>(bias, W, corr);

    dim3 grid(DOUT / 64, NT / 64);
    moe_wmma_kernel<<<grid, 256, 0, stream>>>(Abf, Wbf, weights, corr, multi, expo);
}
